// GraphAttentionLayer_90108413870267
// MI455X (gfx1250) — matrix operand order repaired and checked by reading, not compiled, not run on hardware
//
#include <hip/hip_runtime.h>
#include <hip/hip_bf16.h>

typedef __attribute__((ext_vector_type(16))) __bf16        v16bf;
typedef __attribute__((ext_vector_type(8)))  float         v8f;
typedef __attribute__((ext_vector_type(8)))  unsigned int  v8u;
typedef __attribute__((ext_vector_type(4)))  unsigned int  v4u;
typedef __attribute__((ext_vector_type(4)))  float         v4f;
typedef __attribute__((ext_vector_type(4)))  int           v4i;

#define N_NODES   6144
#define IN_DIM    256
#define OUT_DIM   256
#define NUM_HEADS 4
#define HEAD_DIM  64
#define NEG_SLOPE 0.2f
#define JCHUNKS   (N_NODES / 32)   // 192
#define NDW       (N_NODES / 2)    // 3072 packed-bf16 dwords per adj row
#define LOG2E     1.44269504088896340736f

union U16bf { v8u u; v16bf b; };

// RNE f32 -> bf16 (cold paths only)
__device__ __forceinline__ unsigned int f2bf(float f) {
  unsigned int u = __float_as_uint(f);
  unsigned int r = u + 0x7FFFu + ((u >> 16) & 1u);
  return r >> 16;
}
__device__ __forceinline__ unsigned int pack_bf16(float lo, float hi) {
  return f2bf(lo) | (f2bf(hi) << 16);
}

__device__ __forceinline__ float fast_exp2(float x) {
#if __has_builtin(__builtin_amdgcn_exp2f)
  return __builtin_amdgcn_exp2f(x);
#else
  return exp2f(x);
#endif
}
__device__ __forceinline__ float fast_rcp(float x) {
#if __has_builtin(__builtin_amdgcn_rcpf)
  return __builtin_amdgcn_rcpf(x);
#else
  return 1.0f / x;
#endif
}

// packed bf16 ops (no clang builtins for these -> inline CDNA5 asm)
__device__ __forceinline__ unsigned int pk_mul_bf16(unsigned int a, unsigned int b) {
  unsigned int d;
  asm("v_pk_mul_bf16 %0, %1, %2" : "=v"(d) : "v"(a), "v"(b));
  return d;
}
__device__ __forceinline__ unsigned int pk_max_bf16(unsigned int a, unsigned int b) {
  unsigned int d;
  asm("v_pk_max_num_bf16 %0, %1, %2" : "=v"(d) : "v"(a), "v"(b));
  return d;
}

// two softmax weights (packed bf16):
//   p = mask * max(Es*Ed, Ecs*Ed2)     (= mask * exp2(max(u,t)), exp2 monotone)
__device__ __forceinline__ unsigned int pelem(unsigned int ed, unsigned int ed2,
                                              unsigned int es, unsigned int ecs,
                                              unsigned int am) {
  return pk_mul_bf16(pk_max_bf16(pk_mul_bf16(es, ed), pk_mul_bf16(ecs, ed2)), am);
}

// ---------------------------------------------------------------------------
// Kernel A: Wh = x @ W   (6144x256 @ 256x256, f32)
// ---------------------------------------------------------------------------
__global__ __launch_bounds__(256) void wh_kernel(const float* __restrict__ x,
                                                 const float* __restrict__ W,
                                                 float* __restrict__ Wh) {
  __shared__ __align__(16) float xs[IN_DIM];
  int row = blockIdx.x;
  int t = threadIdx.x;
  xs[t] = x[(size_t)row * IN_DIM + t];
  __syncthreads();
  float sum = 0.f;
#pragma unroll 8
  for (int k = 0; k < IN_DIM; ++k)
    sum = fmaf(xs[k], W[k * OUT_DIM + t], sum);
  Wh[(size_t)row * OUT_DIM + t] = sum;
}

// ---------------------------------------------------------------------------
// Kernel B: f_src[h][n], f_dst[h][n] = dot(Wh[n,h,:], a_src/a_dst)
// ---------------------------------------------------------------------------
__global__ __launch_bounds__(256) void fvec_kernel(const float* __restrict__ Wh,
                                                   const float* __restrict__ a,
                                                   float* __restrict__ fsrc,
                                                   float* __restrict__ fdst) {
  __shared__ __align__(16) float as[2 * HEAD_DIM];
  int t = threadIdx.x;
  if (t < 2 * HEAD_DIM) as[t] = a[t];
  __syncthreads();
  int tid = blockIdx.x * 256 + t;
  int h = tid / N_NODES;
  int n = tid - h * N_NODES;
  const v4f* wv = (const v4f*)(Wh + (size_t)n * OUT_DIM + h * HEAD_DIM);
  float s = 0.f, d = 0.f;
#pragma unroll
  for (int k4 = 0; k4 < HEAD_DIM / 4; ++k4) {
    v4f v = wv[k4];
#pragma unroll
    for (int j = 0; j < 4; ++j) {
      s = fmaf(v[j], as[k4 * 4 + j], s);
      d = fmaf(v[j], as[HEAD_DIM + k4 * 4 + j], d);
    }
  }
  fsrc[h * N_NODES + n] = s;
  fdst[h * N_NODES + n] = d;
}

// ---------------------------------------------------------------------------
// Kernel C: maxf[h] = max_j fdst[h][j]   (single block)
// ---------------------------------------------------------------------------
__global__ __launch_bounds__(256) void maxf_kernel(const float* __restrict__ fdst,
                                                   float* __restrict__ maxf) {
  __shared__ float red[256];
  int t = threadIdx.x;
  int h = t >> 6;
  int i0 = t & 63;
  float m = -3.4e38f;
  for (int i = i0; i < N_NODES; i += 64) m = fmaxf(m, fdst[h * N_NODES + i]);
  red[t] = m;
  __syncthreads();
  for (int s = 32; s >= 1; s >>= 1) {
    if ((t & 63) < s) red[t] = fmaxf(red[t], red[t + s]);
    __syncthreads();
  }
  if ((t & 63) == 0) maxf[h] = red[t];
}

// ---------------------------------------------------------------------------
// Kernel D: pack Wh into bf16 WMMA-B-layout tiles (32x16: N=lane&15;
// lanes 0-15 hold K=0..7,16..23, lanes 16-31 hold K=8..15,24..31, 2 K per dword).
// ---------------------------------------------------------------------------
__global__ __launch_bounds__(128) void pack_kernel(const float* __restrict__ Wh,
                                                   unsigned int* __restrict__ Bpack) {
  int bx = blockIdx.x;               // h*JCHUNKS + jc
  int h = bx / JCHUNKS;
  int jc = bx - h * JCHUNKS;
  int tid = threadIdx.x;
  int tt = tid >> 5;                 // d-tile 0..3
  int lane = tid & 31;
  int kbase = (lane & 16) >> 1;
  int d = h * HEAD_DIM + tt * 16 + (lane & 15);
  int jb = jc * 32 + kbase;
  unsigned int w[8];
#pragma unroll
  for (int r = 0; r < 8; ++r) {
    float lo = Wh[(size_t)(jb + 2 * r + (r >> 2) * 8) * OUT_DIM + d];
    float hi = Wh[(size_t)(jb + 2 * r + (r >> 2) * 8 + 1) * OUT_DIM + d];
    w[r] = pack_bf16(lo, hi);
  }
  unsigned int* dst = Bpack + ((size_t)bx * 4 + tt) * 256 + lane * 8;
  *(v4u*)(dst + 0) = (v4u){w[0], w[1], w[2], w[3]};
  *(v4u*)(dst + 4) = (v4u){w[4], w[5], w[6], w[7]};
}

// ---------------------------------------------------------------------------
// Kernel E: adj (int32 0/1) -> packed bf16 {0.0,1.0} pairs (halves adj BW)
// ---------------------------------------------------------------------------
__global__ __launch_bounds__(256) void adjbf_kernel(const int* __restrict__ adj,
                                                    unsigned int* __restrict__ adjbf) {
  size_t i = (size_t)blockIdx.x * 256 + threadIdx.x;   // one v4u (8 ints) each
  const v4i* src = (const v4i*)adj;
  v4i a = src[2 * i];
  v4i b = src[2 * i + 1];
  v4u o;
  o[0] = (a[0] ? 0x3F80u : 0u) | (a[1] ? 0x3F800000u : 0u);
  o[1] = (a[2] ? 0x3F80u : 0u) | (a[3] ? 0x3F800000u : 0u);
  o[2] = (b[0] ? 0x3F80u : 0u) | (b[1] ? 0x3F800000u : 0u);
  o[3] = (b[2] ? 0x3F80u : 0u) | (b[3] ? 0x3F800000u : 0u);
  ((v4u*)adjbf)[i] = o;
}

// ---------------------------------------------------------------------------
// Main kernel: fused masked-softmax attention, single pass over adjbf.
// p = mask * max(Es*Ed[j], Ecs*Ed2[j]) entirely in packed bf16; rowsums are
// produced by an extra WMMA against an all-ones B tile (denominator lands in
// exactly the C-layout needed for normalization -> no LDS reduce, no barrier).
// Block = 128 threads = 4 waves; wave owns 32 rows (two A tiles share B tiles
// and score tables) -> 10 WMMAs per 32-j chunk.
// ---------------------------------------------------------------------------
__global__ __launch_bounds__(128) void gat_attn_kernel(
    const unsigned int* __restrict__ adjbf, const float* __restrict__ fsrc,
    const float* __restrict__ fdst, const float* __restrict__ maxf,
    const unsigned int* __restrict__ Bpack, float* __restrict__ out) {
  __shared__ __align__(16) unsigned int edp[NDW];   // packed bf16 exp2(fd2)
  __shared__ __align__(16) unsigned int e2p[NDW];   // packed bf16 exp2(0.2*fd2)

  int bx = blockIdx.x;
  int head = bx & 3;                 // heads adjacent -> adjbf rows reused in L2
  int rb = bx >> 2;                  // 128-row block, 0..47
  int tid = threadIdx.x;
  int wave = tid >> 5;
  int lane = tid & 31;

  // stage per-column score tables in LDS (24 KB)
  {
    const float* fd = fdst + head * N_NODES;
    for (int i = tid; i < NDW; i += 128) {
      float f0 = fd[2 * i] * LOG2E;
      float f1 = fd[2 * i + 1] * LOG2E;
      edp[i] = pack_bf16(fast_exp2(f0), fast_exp2(f1));
      e2p[i] = pack_bf16(fast_exp2(NEG_SLOPE * f0), fast_exp2(NEG_SLOPE * f1));
    }
  }
  __syncthreads();

  int hi = lane >> 4;
  int m = lane & 15;
  int kd = hi * 4;                   // dword offset of this lane's K half
  int row0 = rb * 128 + wave * 32 + m;          // tile a
  int row1 = row0 + 16;                         // tile b
  float mx = maxf[head];
  float fs0 = fsrc[head * N_NODES + row0];
  float fs1 = fsrc[head * N_NODES + row1];
  float fm0 = fs0 + mx, fm1 = fs1 + mx;
  float Mi0 = (fm0 > 0.f ? fm0 : NEG_SLOPE * fm0) * LOG2E;  // row upper bound (log2)
  float Mi1 = (fm1 > 0.f ? fm1 : NEG_SLOPE * fm1) * LOG2E;
  float Es0  = fast_exp2(fmaf(fs0, LOG2E, -Mi0));
  float Es1  = fast_exp2(fmaf(fs1, LOG2E, -Mi1));
  float Ecs0 = fast_exp2(fmaf(fs0, NEG_SLOPE * LOG2E, -Mi0));
  float Ecs1 = fast_exp2(fmaf(fs1, NEG_SLOPE * LOG2E, -Mi1));
  unsigned int es0  = pack_bf16(Es0, Es0),   es1  = pack_bf16(Es1, Es1);
  unsigned int ecs0 = pack_bf16(Ecs0, Ecs0), ecs1 = pack_bf16(Ecs1, Ecs1);

  v8f acc0a = {}, acc1a = {}, acc2a = {}, acc3a = {}, accSa = {};
  v8f acc0b = {}, acc1b = {}, acc2b = {}, acc3b = {}, accSb = {};
  const unsigned int* ab0 = adjbf + (size_t)row0 * NDW;
  const unsigned int* ab1 = adjbf + (size_t)row1 * NDW;
  const unsigned int* bhead = Bpack + (size_t)head * JCHUNKS * 1024 + lane * 8;
  U16bf Bones;
  Bones.u = (v8u){0x3F803F80u, 0x3F803F80u, 0x3F803F80u, 0x3F803F80u,
                  0x3F803F80u, 0x3F803F80u, 0x3F803F80u, 0x3F803F80u};

  for (int jc = 0; jc < JCHUNKS; ++jc) {
    int d0 = jc * 16 + kd;           // group0 dwords (K k0..k0+7)
    int d1 = d0 + 8;                 // group1 dwords (K 16+k0 ..)
    // speculative prefetch 8 chunks ahead (global_prefetch_b8)
    __builtin_prefetch(ab0 + d0 + 128, 0, 1);
    __builtin_prefetch(ab1 + d0 + 128, 0, 1);

    v4u m0a = *(const v4u*)(ab0 + d0);
    v4u m1a = *(const v4u*)(ab0 + d1);
    v4u m0b = *(const v4u*)(ab1 + d0);
    v4u m1b = *(const v4u*)(ab1 + d1);
    v4u ed0 = *(const v4u*)(edp + d0);
    v4u ed1 = *(const v4u*)(edp + d1);
    v4u e20 = *(const v4u*)(e2p + d0);
    v4u e21 = *(const v4u*)(e2p + d1);

    U16bf A0, A1;
#pragma unroll
    for (int r = 0; r < 4; ++r) {
      A0.u[r]     = pelem(ed0[r], e20[r], es0, ecs0, m0a[r]);
      A0.u[4 + r] = pelem(ed1[r], e21[r], es0, ecs0, m1a[r]);
      A1.u[r]     = pelem(ed0[r], e20[r], es1, ecs1, m0b[r]);
      A1.u[4 + r] = pelem(ed1[r], e21[r], es1, ecs1, m1b[r]);
    }

    const unsigned int* bb = bhead + (size_t)jc * 1024;
    U16bf B0, B1, B2, B3;
    B0.u = *(const v8u*)(bb + 0);
    B1.u = *(const v8u*)(bb + 256);
    B2.u = *(const v8u*)(bb + 512);
    B3.u = *(const v8u*)(bb + 768);

    acc0a = __builtin_amdgcn_wmma_f32_16x16x32_bf16(false, A0.b, false, B0.b, (short)0, acc0a, false, false);
    acc0b = __builtin_amdgcn_wmma_f32_16x16x32_bf16(false, A1.b, false, B0.b, (short)0, acc0b, false, false);
    acc1a = __builtin_amdgcn_wmma_f32_16x16x32_bf16(false, A0.b, false, B1.b, (short)0, acc1a, false, false);
    acc1b = __builtin_amdgcn_wmma_f32_16x16x32_bf16(false, A1.b, false, B1.b, (short)0, acc1b, false, false);
    acc2a = __builtin_amdgcn_wmma_f32_16x16x32_bf16(false, A0.b, false, B2.b, (short)0, acc2a, false, false);
    acc2b = __builtin_amdgcn_wmma_f32_16x16x32_bf16(false, A1.b, false, B2.b, (short)0, acc2b, false, false);
    acc3a = __builtin_amdgcn_wmma_f32_16x16x32_bf16(false, A0.b, false, B3.b, (short)0, acc3a, false, false);
    acc3b = __builtin_amdgcn_wmma_f32_16x16x32_bf16(false, A1.b, false, B3.b, (short)0, acc3b, false, false);
    // rowsum (softmax denominator) via ones-tile WMMA
    accSa = __builtin_amdgcn_wmma_f32_16x16x32_bf16(false, A0.b, false, Bones.b, (short)0, accSa, false, false);
    accSb = __builtin_amdgcn_wmma_f32_16x16x32_bf16(false, A1.b, false, Bones.b, (short)0, accSb, false, false);
  }

  // accS[r] already holds the row sum for row M=r+8*hi in every lane's column.
  float* orow_base = out + (size_t)(rb * 128 + wave * 32) * OUT_DIM + head * HEAD_DIM;
#pragma unroll
  for (int r = 0; r < 8; ++r) {
    int ra = hi * 8 + r;
    float la = accSa[r], lb = accSb[r];
    float inva = la > 0.f ? fast_rcp(la) : 0.f;   // all-masked row -> 0
    float invb = lb > 0.f ? fast_rcp(lb) : 0.f;
    float* orowa = orow_base + (size_t)ra * OUT_DIM;
    float* orowb = orow_base + (size_t)(16 + ra) * OUT_DIM;
    orowa[0 * 16 + m] = acc0a[r] * inva;
    orowa[1 * 16 + m] = acc1a[r] * inva;
    orowa[2 * 16 + m] = acc2a[r] * inva;
    orowa[3 * 16 + m] = acc3a[r] * inva;
    orowb[0 * 16 + m] = acc0b[r] * invb;
    orowb[1 * 16 + m] = acc1b[r] * invb;
    orowb[2 * 16 + m] = acc2b[r] * invb;
    orowb[3 * 16 + m] = acc3b[r] * invb;
  }
}

// ---------------------------------------------------------------------------
// Launcher
// ---------------------------------------------------------------------------
extern "C" void kernel_launch(void* const* d_in, const int* in_sizes, int n_in,
                              void* d_out, int out_size, void* d_ws, size_t ws_size,
                              hipStream_t stream) {
  (void)in_sizes; (void)n_in; (void)out_size; (void)ws_size;
  const float* x   = (const float*)d_in[0];
  const int*   adj = (const int*)d_in[1];
  const float* W   = (const float*)d_in[2];
  const float* a   = (const float*)d_in[3];
  float* out = (float*)d_out;

  char* ws = (char*)d_ws;
  unsigned int* adjbf = (unsigned int*)(ws);              // 75,497,472 B
  float*        Wh    = (float*)(ws + 75497472);          //  6,291,456 B
  unsigned int* Bpack = (unsigned int*)(ws + 81788928);   //  3,145,728 B
  float*        fsrc  = (float*)(ws + 84934656);          //     98,304 B
  float*        fdst  = (float*)(ws + 85032960);          //     98,304 B
  float*        maxf  = (float*)(ws + 85131264);          //         16 B

  adjbf_kernel<<<(N_NODES * (N_NODES / 8)) / 256, 256, 0, stream>>>(adj, adjbf);
  wh_kernel<<<N_NODES, 256, 0, stream>>>(x, W, Wh);
  fvec_kernel<<<NUM_HEADS * N_NODES / 256, 256, 0, stream>>>(Wh, a, fsrc, fdst);
  maxf_kernel<<<1, 256, 0, stream>>>(fdst, maxf);
  pack_kernel<<<NUM_HEADS * JCHUNKS, 128, 0, stream>>>(Wh, Bpack);
  gat_attn_kernel<<<NUM_HEADS * (N_NODES / 128), 128, 0, stream>>>(adjbf, fsrc, fdst, maxf, Bpack, out);
}
